// Model_22797686407346
// MI455X (gfx1250) — matrix operand order repaired and checked by reading, not compiled, not run on hardware
//
#include <hip/hip_runtime.h>

typedef __attribute__((ext_vector_type(16))) _Float16 v16h;
typedef __attribute__((ext_vector_type(8)))  float    v8f;
typedef int vi4 __attribute__((vector_size(16)));   // matches builtin's pointee type

#define B_   64
#define F_   80
#define T_   201
#define HID_ 256

// ---------------------------------------------------------------------------
// Async global -> LDS staging (CDNA5 GLOBAL_LOAD_ASYNC_TO_LDS_B128, ASYNCcnt).
// Guarded by __has_builtin so the module always compiles; fallback is a
// synchronous vector copy (no wait needed in that config).
// ---------------------------------------------------------------------------
#if __has_builtin(__builtin_amdgcn_global_load_async_to_lds_b128)
#define HAVE_ASYNC_LDS 1
#else
#define HAVE_ASYNC_LDS 0
#endif

__device__ __forceinline__ void async_copy_b128(const _Float16* gsrc, _Float16* lds) {
#if HAVE_ASYNC_LDS
  __builtin_amdgcn_global_load_async_to_lds_b128(
      (__attribute__((address_space(1))) vi4*)(void*)gsrc,
      (__attribute__((address_space(3))) vi4*)(void*)lds, 0, 0);
#else
  *(uint4*)lds = *(const uint4*)gsrc;
#endif
}

__device__ __forceinline__ void async_wait_all() {
#if HAVE_ASYNC_LDS
#if __has_builtin(__builtin_amdgcn_s_wait_asynccnt)
  __builtin_amdgcn_s_wait_asynccnt(0);
#else
  asm volatile("s_wait_asynccnt 0" ::: "memory");
#endif
#endif
}

// ---------------------------------------------------------------------------
// WMMA helpers (CDNA5 gfx1250, wave32). D = A(16x32 f16) * B(32x16 f16) + C(f32)
// ---------------------------------------------------------------------------
__device__ __forceinline__ v8f wmma_f16(v16h a, v16h b, v8f c) {
  return __builtin_amdgcn_wmma_f32_16x16x32_f16(
      /*neg_a=*/false, a, /*neg_b=*/false, b,
      /*c_mod=*/(short)0, c, /*reuse_a=*/false, /*reuse_b=*/false);
}

// A-matrix 16x32 f16 fragment loader (ISA 7.12.2 layout):
// lane l: M = l%16, K-group = l/16 (adds 8 to K).
// VGPR v (v<4): K = 2v,2v+1 ; (v>=4): K = 16+2(v-4), +1.  Pairs loaded as u32
// (compiler merges the two contiguous 16B runs into ds/global b128 loads).
__device__ __forceinline__ v16h load_a_frag(const _Float16* p, int rowStride, int lane) {
  const int m = lane & 15;
  const int g = (lane >> 4) * 8;
  union { v16h v; unsigned u[8]; } r;
  const _Float16* row = p + (size_t)m * rowStride + g;
#pragma unroll
  for (int v = 0; v < 8; ++v) {
    const int kl = (v < 4) ? (2 * v) : (16 + 2 * (v - 4));
    r.u[v] = *(const unsigned*)(row + kl);
  }
  return r.v;
}

// B fragments are pre-swizzled: 16 contiguous halves per lane per (ntile, kstep).
__device__ __forceinline__ v16h load_b_frag(const _Float16* sw, int ntile, int kk, int lane) {
  const _Float16* p = sw + ((((size_t)ntile * 16 + kk) * 32 + lane) * 16);
  union { v16h v; uint4 q[2]; } r;
  r.q[0] = *(const uint4*)(p);
  r.q[1] = *(const uint4*)(p + 8);
  return r.v;
}

// ---------------------------------------------------------------------------
// Weight swizzle: row-major f32 W[K][N] -> per-lane B-fragment f16 stream.
// B layout (ISA 7.12.2): lane l: N = l%16, K-group = l/16 (adds 8 to K);
// VGPR v (v<4): K = group*8 + 2v, 2v+1 ; (v>=4): K = 16 + group*8 + 2(v-4), +1.
// ---------------------------------------------------------------------------
__global__ void swizzle_b_kernel(const float* __restrict__ src, _Float16* __restrict__ dst,
                                 int N, int ksteps, int total) {
  int idx = blockIdx.x * 256 + threadIdx.x;
  if (idx >= total) return;
  int i     = idx & 15;
  int lane  = (idx >> 4) & 31;
  int kk    = (idx >> 9) % ksteps;
  int ntile = idx / (ksteps << 9);
  int n = ntile * 16 + (lane & 15);
  int g = lane >> 4;
  int v = i >> 1, w = i & 1;
  int k = kk * 32 + ((v < 4) ? (g * 8 + v * 2 + w) : (16 + g * 8 + (v - 4) * 2 + w));
  dst[idx] = (_Float16)src[(size_t)k * N + n];
}

// ---------------------------------------------------------------------------
// conv1 (8x8 SAME, 1->32) + relu + maxpool(5,1) fused.  Output f16 [B,16,T,32].
// ---------------------------------------------------------------------------
__global__ void conv1_pool_kernel(const float* __restrict__ x, const float* __restrict__ w1,
                                  const float* __restrict__ b1, _Float16* __restrict__ p1) {
  int idx = blockIdx.x * 256 + threadIdx.x;
  if (idx >= B_ * 16 * T_ * 32) return;
  int c  = idx & 31;
  int t  = (idx >> 5) % T_;
  int hp = (idx / (32 * T_)) & 15;
  int b  = idx / (32 * T_ * 16);
  float mx = -3.0e38f;
  for (int hh = 0; hh < 5; ++hh) {
    int h = hp * 5 + hh;
    float acc = b1[c];
#pragma unroll
    for (int kh = 0; kh < 8; ++kh) {
      int hi = h + kh - 3;                 // SAME pad_lo = 3
      if ((unsigned)hi >= (unsigned)F_) continue;
#pragma unroll
      for (int kw = 0; kw < 8; ++kw) {
        int ti = t + kw - 3;
        if ((unsigned)ti >= (unsigned)T_) continue;
        acc += x[((size_t)b * F_ + hi) * T_ + ti] * w1[(kh * 8 + kw) * 32 + c];
      }
    }
    mx = fmaxf(mx, acc);
  }
  p1[idx] = (_Float16)fmaxf(mx, 0.f);     // relu(max) == max(relu)
}

// ---------------------------------------------------------------------------
// conv2 (4x4 SAME, 32->64) via im2col + WMMA, fused relu + maxpool(4,1).
// Writes RNN input xs0 f16 in [T][B][256] order, feature d = co*4 + hp.
// Grid: (ceil(T/16), B), block 128 (4 waves; wave w = N-tile w).
// ---------------------------------------------------------------------------
__global__ __launch_bounds__(128)
void conv2_pool_kernel(const _Float16* __restrict__ p1, const _Float16* __restrict__ w2_sw,
                       const float* __restrict__ b2, _Float16* __restrict__ xs0) {
  __shared__ _Float16 a_lds[16 * 512];
  const int tid   = threadIdx.x;
  const int lane  = tid & 31;
  const int ntile = tid >> 5;           // 0..3
  const int t0    = blockIdx.x * 16;
  const int b     = blockIdx.y;
  const int n     = lane & 15;
  const int mhi   = (lane >> 4) * 8;
  const int co    = ntile * 16 + n;
  const float bias = b2[co];

  for (int hp = 0; hp < 4; ++hp) {
    v8f pool;
#pragma unroll
    for (int j = 0; j < 8; ++j) pool[j] = -3.0e38f;
    for (int hh = 0; hh < 4; ++hh) {
      const int h = hp * 4 + hh;
      __syncthreads();
      // im2col: k = (kh*4+kw)*32 + ci ; zero-pad out of range
      for (int idx = tid; idx < 16 * 512; idx += 128) {
        int m = idx >> 9;
        int k = idx & 511;
        int kh = k >> 7, kw = (k >> 5) & 3, ci = k & 31;
        int hi = h + kh - 1;               // SAME pad_lo = 1
        int ti = t0 + m + kw - 1;
        _Float16 vv = (_Float16)0.f;
        if ((unsigned)hi < 16u && (unsigned)ti < (unsigned)T_)
          vv = p1[(((size_t)b * 16 + hi) * T_ + ti) * 32 + ci];
        a_lds[idx] = vv;
      }
      __syncthreads();
      v8f acc = {};
      for (int kk = 0; kk < 16; ++kk) {
        v16h a  = load_a_frag(&a_lds[kk * 32], 512, lane);
        v16h bf = load_b_frag(w2_sw, ntile, kk, lane);
        acc = wmma_f16(a, bf, acc);
      }
#pragma unroll
      for (int j = 0; j < 8; ++j) pool[j] = fmaxf(pool[j], acc[j]);
    }
#pragma unroll
    for (int j = 0; j < 8; ++j) {
      int m = j + mhi;
      int t = t0 + m;
      if (t < T_) {
        float val = fmaxf(pool[j] + bias, 0.f);
        xs0[((size_t)t * B_ + b) * HID_ + co * 4 + hp] = (_Float16)val;
      }
    }
  }
}

// ---------------------------------------------------------------------------
// GRU layer (TF GRUCell). Batch split over 4 blocks x 16 rows; 16 waves/block,
// wave nt owns the 16x16 tile (batch rows x hidden cols nt*16..+15).
// x[t] is double-buffered in LDS via async global->LDS b128 copies issued one
// step ahead (ASYNCcnt), so both GEMM phases read A fragments purely from LDS.
// ---------------------------------------------------------------------------
__global__ __launch_bounds__(512)
void gru_layer_kernel(const _Float16* __restrict__ x16,
                      const _Float16* __restrict__ wg_sw, const float* __restrict__ gb,
                      const _Float16* __restrict__ wc_sw, const float* __restrict__ cb,
                      _Float16* __restrict__ out16, float* __restrict__ out32) {
  __shared__ _Float16 x_buf[2][16 * HID_];
  __shared__ _Float16 h_lds [16 * HID_];
  __shared__ _Float16 rh_lds[16 * HID_];
  const int tid     = threadIdx.x;
  const int lane    = tid & 31;
  const int nt      = tid >> 5;          // wave id == hidden N-tile, 0..15
  const int rowbase = blockIdx.x * 16;   // batch rows handled by this block
  const int n       = lane & 15;
  const int mhi     = (lane >> 4) * 8;
  const int col     = nt * 16 + n;

  const float biasR = gb[col];
  const float biasU = gb[HID_ + col];
  const float biasC = cb[col];

  // Warm L2 with this wave's loop-invariant weight fragment streams.
  __builtin_prefetch(wg_sw + ((size_t)nt * 16) * 32 * 16, 0, 1);
  __builtin_prefetch(wg_sw + ((size_t)(16 + nt) * 16) * 32 * 16, 0, 1);
  __builtin_prefetch(wc_sw + ((size_t)nt * 16) * 32 * 16, 0, 1);

  v8f h = {};
  for (int i = tid; i < 16 * HID_; i += 512) h_lds[i] = (_Float16)0.f;

  // Stage x[0] into buffer 0 (one b128 per thread: 512 * 16B = 8KB tile).
  async_copy_b128(x16 + (size_t)rowbase * HID_ + tid * 8, &x_buf[0][tid * 8]);
  async_wait_all();
  __syncthreads();

  for (int t = 0; t < T_; ++t) {
    // Kick off async staging of x[t+1] while we compute on x[t].
    if (t + 1 < T_) {
      async_copy_b128(x16 + ((size_t)(t + 1) * B_ + rowbase) * HID_ + tid * 8,
                      &x_buf[(t + 1) & 1][tid * 8]);
    }
    const _Float16* xrow = &x_buf[t & 1][0];

    // ---- gates: ru = sigmoid([x,h] @ gk + gb) ------------------------------
    v8f racc = {}, uacc = {};
    for (int kk = 0; kk < 16; ++kk) {
      v16h a = (kk < 8) ? load_a_frag(xrow + kk * 32, HID_, lane)
                        : load_a_frag(&h_lds[(kk - 8) * 32], HID_, lane);
      v16h brf = load_b_frag(wg_sw, nt, kk, lane);        // r columns
      v16h buf = load_b_frag(wg_sw, 16 + nt, kk, lane);   // u columns
      racc = wmma_f16(a, brf, racc);
      uacc = wmma_f16(a, buf, uacc);
    }
    v8f u;
#pragma unroll
    for (int j = 0; j < 8; ++j) {
      float r = 1.f / (1.f + __expf(-(racc[j] + biasR)));
      u[j]    = 1.f / (1.f + __expf(-(uacc[j] + biasU)));
      rh_lds[(j + mhi) * HID_ + col] = (_Float16)(r * h[j]);   // r * h (tile layout)
    }
    __syncthreads();

    // ---- candidate: c = tanh([x, r*h] @ ck + cb) ---------------------------
    v8f cacc = {};
    for (int kk = 0; kk < 16; ++kk) {
      v16h a = (kk < 8) ? load_a_frag(xrow + kk * 32, HID_, lane)
                        : load_a_frag(&rh_lds[(kk - 8) * 32], HID_, lane);
      v16h bcf = load_b_frag(wc_sw, nt, kk, lane);
      cacc = wmma_f16(a, bcf, cacc);
    }
    const size_t obase = ((size_t)t * B_ + rowbase) * HID_;
#pragma unroll
    for (int j = 0; j < 8; ++j) {
      float c = tanhf(cacc[j] + biasC);
      h[j] = u[j] * h[j] + (1.f - u[j]) * c;
      _Float16 hf = (_Float16)h[j];
      h_lds[(j + mhi) * HID_ + col] = hf;              // safe: h_lds unread this phase
      out16[obase + (size_t)(j + mhi) * HID_ + col] = hf;
      if (out32) out32[obase + (size_t)(j + mhi) * HID_ + col] = h[j];
    }
    // Drain this wave's async copy of x[t+1], then publish to all waves.
    async_wait_all();
    __syncthreads();
  }
}

// ---------------------------------------------------------------------------
// FC(256->100) relu, FC(100->2), softmax over the 2 channels. Out [B,2,T] f32.
// ---------------------------------------------------------------------------
__global__ void fc_softmax_kernel(const float* __restrict__ y, const float* __restrict__ w1,
                                  const float* __restrict__ b1, const float* __restrict__ w2,
                                  const float* __restrict__ b2, float* __restrict__ out) {
  int idx = blockIdx.x * 256 + threadIdx.x;
  if (idx >= B_ * T_) return;
  int t = idx % T_, b = idx / T_;
  const float* yr = y + ((size_t)t * B_ + b) * HID_;
  float l0 = b2[0], l1 = b2[1];
  for (int j = 0; j < 100; ++j) {
    float hj = b1[j];
    for (int d = 0; d < HID_; ++d) hj += yr[d] * w1[d * 100 + j];
    hj = fmaxf(hj, 0.f);
    l0 += hj * w2[j * 2 + 0];
    l1 += hj * w2[j * 2 + 1];
  }
  float mx = fmaxf(l0, l1);
  float e0 = __expf(l0 - mx), e1 = __expf(l1 - mx), s = e0 + e1;
  out[((size_t)b * 2 + 0) * T_ + t] = e0 / s;
  out[((size_t)b * 2 + 1) * T_ + t] = e1 / s;
}

// ---------------------------------------------------------------------------
extern "C" void kernel_launch(void* const* d_in, const int* in_sizes, int n_in,
                              void* d_out, int out_size, void* d_ws, size_t ws_size,
                              hipStream_t stream) {
  const float* x    = (const float*)d_in[0];
  const float* w1   = (const float*)d_in[1];
  const float* b1   = (const float*)d_in[2];
  const float* w2   = (const float*)d_in[3];
  const float* b2   = (const float*)d_in[4];
  const float* g0k  = (const float*)d_in[5];
  const float* g0b  = (const float*)d_in[6];
  const float* c0k  = (const float*)d_in[7];
  const float* c0b  = (const float*)d_in[8];
  const float* g1k  = (const float*)d_in[9];
  const float* g1b  = (const float*)d_in[10];
  const float* c1k  = (const float*)d_in[11];
  const float* c1b  = (const float*)d_in[12];
  const float* fc1w = (const float*)d_in[13];
  const float* fc1b = (const float*)d_in[14];
  const float* fcfw = (const float*)d_in[15];
  const float* fcfb = (const float*)d_in[16];

  char*  ws  = (char*)d_ws;
  size_t off = 0;
  auto salloc = [&](size_t bytes) -> void* {
    void* p = ws + off;
    off = (off + bytes + 255) & ~(size_t)255;
    return p;
  };
  _Float16* p1    = (_Float16*)salloc((size_t)B_ * 16 * T_ * 32 * 2);  // conv1 pooled
  _Float16* xs0   = (_Float16*)salloc((size_t)T_ * B_ * HID_ * 2);     // RNN input
  _Float16* ys0   = (_Float16*)salloc((size_t)T_ * B_ * HID_ * 2);     // GRU0 out f16
  _Float16* ys1   = (_Float16*)salloc((size_t)T_ * B_ * HID_ * 2);     // GRU1 out f16
  float*    ys1f  = (float*)   salloc((size_t)T_ * B_ * HID_ * 4);     // GRU1 out f32
  _Float16* w2sw  = (_Float16*)salloc((size_t)4  * 16 * 512 * 2);      // conv2 W frag
  _Float16* g0sw  = (_Float16*)salloc((size_t)32 * 16 * 512 * 2);
  _Float16* c0sw  = (_Float16*)salloc((size_t)16 * 16 * 512 * 2);
  _Float16* g1sw  = (_Float16*)salloc((size_t)32 * 16 * 512 * 2);
  _Float16* c1sw  = (_Float16*)salloc((size_t)16 * 16 * 512 * 2);
  (void)ws_size; (void)in_sizes; (void)n_in; (void)out_size;

  int tot;
  tot = 4 * 16 * 512;   // conv2_w: K=512 rows ((kh*4+kw)*32+ci), N=64
  swizzle_b_kernel<<<(tot + 255) / 256, 256, 0, stream>>>(w2, w2sw, 64, 16, tot);
  tot = 32 * 16 * 512;
  swizzle_b_kernel<<<(tot + 255) / 256, 256, 0, stream>>>(g0k, g0sw, 512, 16, tot);
  tot = 16 * 16 * 512;
  swizzle_b_kernel<<<(tot + 255) / 256, 256, 0, stream>>>(c0k, c0sw, 256, 16, tot);
  tot = 32 * 16 * 512;
  swizzle_b_kernel<<<(tot + 255) / 256, 256, 0, stream>>>(g1k, g1sw, 512, 16, tot);
  tot = 16 * 16 * 512;
  swizzle_b_kernel<<<(tot + 255) / 256, 256, 0, stream>>>(c1k, c1sw, 256, 16, tot);

  tot = B_ * 16 * T_ * 32;
  conv1_pool_kernel<<<(tot + 255) / 256, 256, 0, stream>>>(x, w1, b1, p1);

  conv2_pool_kernel<<<dim3((T_ + 15) / 16, B_), 128, 0, stream>>>(p1, w2sw, b2, xs0);

  gru_layer_kernel<<<4, 512, 0, stream>>>(xs0, g0sw, g0b, c0sw, c0b, ys0, nullptr);
  gru_layer_kernel<<<4, 512, 0, stream>>>(ys0, g1sw, g1b, c1sw, c1b, ys1, ys1f);

  tot = B_ * T_;
  fc_softmax_kernel<<<(tot + 255) / 256, 256, 0, stream>>>(ys1f, fc1w, fc1b, fcfw, fcfb,
                                                           (float*)d_out);
}
